// RecurrentApproximator_13159779794961
// MI455X (gfx1250) — matrix operand order repaired and checked by reading, not compiled, not run on hardware
//
#include <hip/hip_runtime.h>

// ---------------------------------------------------------------------------
// Problem constants (match the JAX reference)
// ---------------------------------------------------------------------------
#define Hdim 512
#define NLay 8
#define Tlen 512
#define Bsz  64
#define Llen 512
#define BOSV (-10001.0f)

#define ENC_BLOCKS 16   // 16*256 threads = 128 waves = 128 tiles of [64x512]
#define DEC_BLOCKS 64   // 64*256 threads = 512 waves

// LDS row pitch for staged Whh: 512 elements + 8 pad (16B) -> de-banked rows
#define WPITCH 520

typedef __bf16 bf16_t;
typedef __attribute__((ext_vector_type(16))) __bf16 v16bf;
typedef __attribute__((ext_vector_type(8)))  __bf16 v8bf;
typedef __attribute__((ext_vector_type(8)))  float  v8f;
typedef __attribute__((ext_vector_type(4)))  unsigned u32x4;
typedef __attribute__((ext_vector_type(8)))  int      i32x8;
typedef __attribute__((ext_vector_type(4)))  int      i32x4;

// ---------------------------------------------------------------------------
// Helpers
// ---------------------------------------------------------------------------
__device__ __forceinline__ bf16_t f2bf(float f) {
  unsigned u = __builtin_bit_cast(unsigned, f);
  unsigned r = u + 0x7FFFu + ((u >> 16) & 1u);   // round-to-nearest-even
  unsigned short h = (unsigned short)(r >> 16);
  return __builtin_bit_cast(bf16_t, h);
}
__device__ __forceinline__ float bf2f(bf16_t b) {
  unsigned short h = __builtin_bit_cast(unsigned short, b);
  unsigned u = ((unsigned)h) << 16;
  return __builtin_bit_cast(float, u);
}

__device__ __forceinline__ float wave_sum(float v) {
#pragma unroll
  for (int m = 16; m >= 1; m >>= 1) v += __shfl_xor(v, m, 32);
  return v;
}
__device__ __forceinline__ float wave_max(float v) {
#pragma unroll
  for (int m = 16; m >= 1; m >>= 1) v = fmaxf(v, __shfl_xor(v, m, 32));
  return v;
}

// Grid-wide split barrier via a monotonically increasing counter in global ws.
// Counter must be memset to 0 before each persistent-kernel launch.
__device__ __forceinline__ void grid_sync(unsigned* bar, unsigned nblk, unsigned epoch) {
  __syncthreads();
  if (threadIdx.x == 0) {
    __threadfence();
    __hip_atomic_fetch_add(bar, 1u, __ATOMIC_ACQ_REL, __HIP_MEMORY_SCOPE_AGENT);
    const unsigned target = nblk * epoch;
    while (__hip_atomic_load(bar, __ATOMIC_ACQUIRE, __HIP_MEMORY_SCOPE_AGENT) < target) {
      __builtin_amdgcn_s_sleep(1);
    }
  }
  __syncthreads();
}

// ---------------------------------------------------------------------------
// TDM: async 1-D tensor load (bf16 elements) global -> LDS with row padding
// (16 B of LDS pad inserted after every 1024 B => LDS row pitch 1040 B).
// Descriptor layout per CDNA5 ISA (D# groups 0/1;
// remaining groups zero for <=2-D tensors). count=1, type=2("image"),
// data_size=2B. This toolchain exposes the 6-arg builtin form.
// ---------------------------------------------------------------------------
__device__ __forceinline__ void tdm_load_padded(unsigned lds_off,
                                                const bf16_t* src,
                                                unsigned n_elems) {
#if __has_builtin(__builtin_amdgcn_tensor_load_to_lds)
  unsigned long long ga = (unsigned long long)(const void*)src;
  u32x4 g0;
  g0.x = 1u;                                          // count=1, user mode
  g0.y = lds_off;                                     // lds_addr (bytes)
  g0.z = (unsigned)(ga & 0xFFFFFFFFull);              // global_addr[31:0]
  g0.w = (unsigned)((ga >> 32) & 0x01FFFFFFull)       // global_addr[56:32]
         | (2u << 30);                                // type=2
  i32x8 g1;
  g1[0] = (int)((1u << 16)      // data_size = 1 -> 2 bytes
                | (1u << 20)    // pad_enable
                | (7u << 22)    // pad_interval = 256 DWORDs (1024 B)
                | (3u << 25));  // pad_amount   = 4 DWORDs  (16 B)
  g1[1] = (int)((n_elems & 0xFFFFu) << 16);           // tensor_dim0[15:0]
  g1[2] = (int)(n_elems >> 16);                       // tensor_dim0[31:16]
  g1[3] = (int)((n_elems & 0xFFFFu) << 16);           // tile_dim0
  g1[4] = 0;                                          // tile_dim1/2 unused (1-D)
  g1[5] = (int)n_elems;                               // tensor_dim0_stride
  g1[6] = 0;
  g1[7] = 0;
  i32x4 gz4 = {0, 0, 0, 0};
  i32x8 gz8 = {0, 0, 0, 0, 0, 0, 0, 0};
  __builtin_amdgcn_tensor_load_to_lds(g0, g1, gz4, gz4, gz8, 0);
  __builtin_amdgcn_s_wait_tensorcnt((short)0);
#else
  (void)lds_off; (void)src; (void)n_elems;  // fallback handled by caller
#endif
}

// ---------------------------------------------------------------------------
// Core WMMA tile accumulator:  C(16x16,f32) += A(16xK,bf16) * W(NxK,bf16)^T
// A row-major offset to (m0,0); W row-major [N x ldb] offset to (n0,0).
// A layout: lane<16 -> K {kb..kb+7, kb+16..kb+23}; lane>=16 -> {+8, +24}
// B layout: lane<16 -> K {kb..kb+7, kb+16..kb+23}; lane>=16 -> {+8, +24}
// Works for global or LDS B operand (address space inferred after inlining).
// ---------------------------------------------------------------------------
__device__ __forceinline__ v8f wmma_acc_bf16(v8f c,
                                             const bf16_t* __restrict__ Arow, int lda,
                                             const bf16_t* __restrict__ Brow, int ldb,
                                             int K, int lane) {
  const int half = lane >> 4;
  const int r    = lane & 15;
  const bf16_t* ap = Arow + (size_t)r * lda + half * 8;
  const bf16_t* bp = Brow + (size_t)r * ldb + half * 8;
  for (int k = 0; k < K; k += 32) {
    v8bf alo = *(const v8bf*)(ap + k);
    v8bf ahi = *(const v8bf*)(ap + k + 16);
    v8bf blo = *(const v8bf*)(bp + k);
    v8bf bhi = *(const v8bf*)(bp + k + 16);
    v16bf a, b;
#pragma unroll
    for (int i = 0; i < 8; ++i) {
      a[i] = alo[i]; a[i + 8] = ahi[i];
      b[i] = blo[i]; b[i + 8] = bhi[i];
    }
    c = __builtin_amdgcn_wmma_f32_16x16x32_bf16(false, a, false, b, (short)0, c,
                                                false, false);
  }
  return c;
}

// ---------------------------------------------------------------------------
// Kernel: f32 -> bf16 bulk conversion (weights, hidden init)
// ---------------------------------------------------------------------------
__global__ __launch_bounds__(256) void k_conv_bf16(const float* __restrict__ s,
                                                   bf16_t* __restrict__ d, int n) {
  int i = blockIdx.x * blockDim.x + threadIdx.x;
  if (i < n) d[i] = f2bf(s[i]);
}

// ---------------------------------------------------------------------------
// Kernel: embedding  x[b,l,h] = inputs[b,l]*We[h] + be[h]  (bf16 out)
// ---------------------------------------------------------------------------
__global__ __launch_bounds__(256) void k_embed(const float* __restrict__ inp,
                                               const float* __restrict__ We,
                                               const float* __restrict__ be,
                                               bf16_t* __restrict__ xbf, int total) {
  int i = blockIdx.x * blockDim.x + threadIdx.x;
  if (i >= total) return;
  int h  = i & (Hdim - 1);
  int bl = i >> 9;               // / Hdim
  xbf[i] = f2bf(inp[bl] * We[h] + be[h]);
}

// ---------------------------------------------------------------------------
// Kernel: Y[m,n] = sum_k X[m,k]*W[n,k] + bias[n]   (bf16 in, f32 out)
// One wave per 16x16 tile.
// ---------------------------------------------------------------------------
__global__ __launch_bounds__(256) void k_gemm_bias(const bf16_t* __restrict__ X,
                                                   const bf16_t* __restrict__ W,
                                                   const float* __restrict__ bias,
                                                   float* __restrict__ Y,
                                                   int M, int N, int K) {
  int wid  = (blockIdx.x * blockDim.x + threadIdx.x) >> 5;
  int lane = threadIdx.x & 31;
  int nt_n = N >> 4;
  if (wid >= (M >> 4) * nt_n) return;
  int mb = wid / nt_n, nb = wid % nt_n;
  v8f c = {};
  c = wmma_acc_bf16(c, X + (size_t)(mb * 16) * K, K,
                       W + (size_t)(nb * 16) * K, K, K, lane);
  int n = nb * 16 + (lane & 15);
  int half = lane >> 4;
  float bn = bias[n];
#pragma unroll
  for (int r = 0; r < 8; ++r) {
    int m = mb * 16 + r + half * 8;
    Y[(size_t)m * N + n] = c[r] + bn;
  }
}

// ---------------------------------------------------------------------------
// Kernel: persistent encoder recurrence for one layer.
//   for t: h = tanh(xw[:,t,:] + h @ Whh^T + bhh); write bf16 to x[:,t,:]
// Wave->tile map: mb = wid&3, nb = wid>>2, so one 8-wave block owns 2 N-tiles
// => 32 rows of Whh staged ONCE into LDS via TDM (reused for all 512 steps).
// ---------------------------------------------------------------------------
__global__ __launch_bounds__(256) void k_enc_scan(const float* __restrict__ xw,
                                                  bf16_t* __restrict__ xbf,
                                                  const bf16_t* __restrict__ Whh,
                                                  const float* __restrict__ bhh,
                                                  bf16_t* __restrict__ hping,
                                                  bf16_t* __restrict__ hpong,
                                                  float* __restrict__ hfinal,
                                                  unsigned* __restrict__ bar) {
  __shared__ __align__(16) bf16_t sWhh[32 * WPITCH];   // ~33 KB

  const int lane = threadIdx.x & 31;
  const int wv   = threadIdx.x >> 5;                            // wave in block
  const int wid  = blockIdx.x * 8 + wv;                         // 0..127
  const int mb   = wid & 3;    // 0..3   (batch tiles)
  const int nb   = wid >> 2;   // 0..31  (hidden tiles)
  const int n    = nb * 16 + (lane & 15);
  const int half = lane >> 4;

  // ---- stage this block's 32 Whh rows into LDS (loop-invariant weights) ----
  const bf16_t* wsrc = Whh + (size_t)blockIdx.x * 32 * Hdim;
#if __has_builtin(__builtin_amdgcn_tensor_load_to_lds)
  if (wv == 0) {
    tdm_load_padded((unsigned)(unsigned long long)&sWhh[0], wsrc, 32 * Hdim);
  }
#else
  for (int i = threadIdx.x; i < 32 * (Hdim / 8); i += blockDim.x) {
    int row = i >> 6, c8 = i & 63;
    *(v8bf*)&sWhh[row * WPITCH + c8 * 8] = *(const v8bf*)&wsrc[row * Hdim + c8 * 8];
  }
#endif
  __syncthreads();

  const bf16_t* sB = &sWhh[(wv >> 2) * 16 * WPITCH];   // this wave's 16 B-rows

  bf16_t* hc = hping;
  bf16_t* hn = hpong;
  unsigned epoch = 0;
  for (int t = 0; t < Llen; ++t) {
    v8f c = {};
    c = wmma_acc_bf16(c, hc + (size_t)(mb * 16) * Hdim, Hdim,
                         sB, WPITCH, Hdim, lane);
    float ht[8];
#pragma unroll
    for (int r = 0; r < 8; ++r) {
      int b = mb * 16 + r + half * 8;
      ht[r] = tanhf(c[r] + xw[((size_t)b * Llen + t) * Hdim + n] + bhh[n]);
    }
    if (t + 1 < Llen) {   // warm L2/L0 for next step's xw column
      int b0 = mb * 16 + half * 8;
      __builtin_prefetch(&xw[((size_t)b0 * Llen + t + 1) * Hdim + n], 0, 0);
    }
    grid_sync(bar, ENC_BLOCKS, ++epoch);   // all reads of hc complete
#pragma unroll
    for (int r = 0; r < 8; ++r) {
      int b = mb * 16 + r + half * 8;
      bf16_t hb = f2bf(ht[r]);
      hn[(size_t)b * Hdim + n] = hb;
      xbf[((size_t)b * Llen + t) * Hdim + n] = hb;
      if (t == Llen - 1) hfinal[(size_t)b * Hdim + n] = ht[r];
    }
    grid_sync(bar, ENC_BLOCKS, ++epoch);   // hn visible before next step
    bf16_t* tmp = hc; hc = hn; hn = tmp;
  }
}

// ---------------------------------------------------------------------------
// Persistent decoder: all T steps, all phases, in one kernel.
// ---------------------------------------------------------------------------
struct DecParams {
  const float*  targets;                 // [B,T]
  const float*  We; const float* be;
  const bf16_t* Wa_bf; const float* ba;
  const float*  Va; const float* bv;
  const float*  ukeys;                   // [B,L,H] f32 (L2-resident)
  const bf16_t* enc_bf;                  // [B,L,H]
  const bf16_t* Wih0_bf;                 // [H,2H]
  const bf16_t* Wih_bf;                  // [7,H,H]
  const bf16_t* Whh_bf;                  // [8,H,H]
  const float*  bih; const float* bhh;   // [8,H]
  float*  hidden_f32;                    // [8,B,H]
  bf16_t* hidden_bf;                     // [8,B,H]
  float*  w_f32;                         // [B,H]
  float*  attn_ws;                       // [B,L]
  bf16_t* inp0_bf;                       // [B,2H]  (emb ++ ctx)
  bf16_t* inpA_bf;                       // [B,H]
  bf16_t* inpB_bf;                       // [B,H]
  float*  top_f32;                       // [B,H]
  float*  out_lp;                        // [B,T,H]
  float*  out_hT;                        // [8,B,H]
  float*  out_attn;                      // [B,T,L]
  unsigned* bar;
};

__global__ __launch_bounds__(256) void k_decoder(DecParams p) {
  const int lane   = threadIdx.x & 31;
  const int wid    = (blockIdx.x * blockDim.x + threadIdx.x) >> 5;  // 0..511
  const int nwaves = (gridDim.x * blockDim.x) >> 5;
  unsigned epoch = 0;

  for (int t = 0; t < Tlen; ++t) {
    // ---- phase 0: w = h[7] @ Wa^T + ba  (waves 0..127) ; embedding (128..191)
    if (wid < 128) {
      int mb = wid >> 5, nb = wid & 31;
      v8f c = {};
      c = wmma_acc_bf16(c, p.hidden_bf + ((size_t)(NLay - 1) * Bsz + mb * 16) * Hdim, Hdim,
                           p.Wa_bf + (size_t)(nb * 16) * Hdim, Hdim, Hdim, lane);
      int n = nb * 16 + (lane & 15), half = lane >> 4;
#pragma unroll
      for (int r = 0; r < 8; ++r) {
        int b = mb * 16 + r + half * 8;
        p.w_f32[(size_t)b * Hdim + n] = c[r] + p.ba[n];
      }
    } else if (wid < 192) {
      int b = wid - 128;
      float iv = (t == 0) ? BOSV : p.targets[(size_t)b * Tlen + (t - 1)];
      for (int h = lane; h < Hdim; h += 32)
        p.inp0_bf[(size_t)b * 2 * Hdim + h] = f2bf(iv * p.We[h] + p.be[h]);
    }
    grid_sync(p.bar, DEC_BLOCKS, ++epoch);

    // ---- phase 1: scores[b,l] = Va . tanh(w[b] + ukeys[b,l]) + bv
    for (int row = wid; row < Bsz * Llen; row += nwaves) {
      int b = row >> 9;            // / L
      const float* uk = p.ukeys + (size_t)row * Hdim;
      const float* wv = p.w_f32 + (size_t)b * Hdim;
      float s = 0.f;
      int h0 = lane * 16;
#pragma unroll
      for (int i = 0; i < 16; ++i) {
        int h = h0 + i;
        s += p.Va[h] * tanhf(wv[h] + uk[h]);
      }
      s = wave_sum(s);
      if (lane == 0) p.attn_ws[row] = s + p.bv[0];
    }
    grid_sync(p.bar, DEC_BLOCKS, ++epoch);

    // ---- phase 2: softmax over L (wave b)
    if (wid < Bsz) {
      float* srow = p.attn_ws + (size_t)wid * Llen;
      float mx = -3.4e38f;
      for (int l = lane; l < Llen; l += 32) mx = fmaxf(mx, srow[l]);
      mx = wave_max(mx);
      float sum = 0.f;
      for (int l = lane; l < Llen; l += 32) sum += __expf(srow[l] - mx);
      sum = wave_sum(sum);
      float inv = 1.f / sum;
      for (int l = lane; l < Llen; l += 32) {
        float a = __expf(srow[l] - mx) * inv;
        srow[l] = a;
        p.out_attn[((size_t)wid * Tlen + t) * Llen + l] = a;
      }
    }
    grid_sync(p.bar, DEC_BLOCKS, ++epoch);

    // ---- phase 3: ctx[b,h] = sum_l attn[b,l]*enc[b,l,h] -> inp0_bf[:,H:2H]
    if (wid < Bsz) {
      int b = wid;
      float acc[16];
#pragma unroll
      for (int i = 0; i < 16; ++i) acc[i] = 0.f;
      int h0 = lane * 16;
      const bf16_t* eb = p.enc_bf + (size_t)b * Llen * Hdim + h0;
      for (int l = 0; l < Llen; ++l) {
        float a = p.attn_ws[(size_t)b * Llen + l];
        const v8bf* ep = (const v8bf*)(eb + (size_t)l * Hdim);
        v8bf e0 = ep[0], e1 = ep[1];
#pragma unroll
        for (int i = 0; i < 8; ++i) {
          acc[i]     += a * bf2f(e0[i]);
          acc[8 + i] += a * bf2f(e1[i]);
        }
      }
#pragma unroll
      for (int i = 0; i < 16; ++i)
        p.inp0_bf[(size_t)b * 2 * Hdim + Hdim + h0 + i] = f2bf(acc[i]);
    }
    grid_sync(p.bar, DEC_BLOCKS, ++epoch);

    // ---- phases 4..: 8 decoder RNN layers (waves 0..127 each own a tile)
    const bf16_t* inp = p.inp0_bf;
    int Kin = 2 * Hdim;
    for (int ly = 0; ly < NLay; ++ly) {
      float hv[8];
      if (wid < 128) {
        int mb = wid >> 5, nb = wid & 31;
        const bf16_t* Wihp = (ly == 0) ? p.Wih0_bf
                                       : p.Wih_bf + (size_t)(ly - 1) * Hdim * Hdim;
        v8f c = {};
        c = wmma_acc_bf16(c, inp + (size_t)(mb * 16) * Kin, Kin,
                             Wihp + (size_t)(nb * 16) * Kin, Kin, Kin, lane);
        c = wmma_acc_bf16(c, p.hidden_bf + ((size_t)ly * Bsz + mb * 16) * Hdim, Hdim,
                             p.Whh_bf + ((size_t)ly * Hdim + nb * 16) * Hdim, Hdim,
                             Hdim, lane);
        int n = nb * 16 + (lane & 15);
        float bias = p.bih[ly * Hdim + n] + p.bhh[ly * Hdim + n];
#pragma unroll
        for (int r = 0; r < 8; ++r) hv[r] = tanhf(c[r] + bias);
      }
      grid_sync(p.bar, DEC_BLOCKS, ++epoch);   // reads of old hidden done
      if (wid < 128) {
        int mb = wid >> 5, nb = wid & 31;
        int n = nb * 16 + (lane & 15), half = lane >> 4;
        bf16_t* nxt = (ly & 1) ? p.inpB_bf : p.inpA_bf;
#pragma unroll
        for (int r = 0; r < 8; ++r) {
          int b = mb * 16 + r + half * 8;
          bf16_t hb = f2bf(hv[r]);
          p.hidden_bf[((size_t)ly * Bsz + b) * Hdim + n]  = hb;
          p.hidden_f32[((size_t)ly * Bsz + b) * Hdim + n] = hv[r];
          if (ly < NLay - 1) nxt[(size_t)b * Hdim + n] = hb;
          else               p.top_f32[(size_t)b * Hdim + n] = hv[r];
        }
      }
      grid_sync(p.bar, DEC_BLOCKS, ++epoch);   // new hidden / input visible
      inp = (ly & 1) ? p.inpB_bf : p.inpA_bf;
      Kin = Hdim;
    }

    // ---- final: log_softmax over H (wave b)
    if (wid < Bsz) {
      const float* xr = p.top_f32 + (size_t)wid * Hdim;
      float mx = -3.4e38f;
      for (int h = lane; h < Hdim; h += 32) mx = fmaxf(mx, xr[h]);
      mx = wave_max(mx);
      float sum = 0.f;
      for (int h = lane; h < Hdim; h += 32) sum += __expf(xr[h] - mx);
      sum = wave_sum(sum);
      float lse = mx + __logf(sum);
      for (int h = lane; h < Hdim; h += 32)
        p.out_lp[((size_t)wid * Tlen + t) * Hdim + h] = xr[h] - lse;
    }
    grid_sync(p.bar, DEC_BLOCKS, ++epoch);
  }

  // final hidden -> d_out hT section
  {
    int total = NLay * Bsz * Hdim;
    int tid   = blockIdx.x * blockDim.x + threadIdx.x;
    int nthr  = gridDim.x * blockDim.x;
    for (int i = tid; i < total; i += nthr) p.out_hT[i] = p.hidden_f32[i];
  }
}

// ---------------------------------------------------------------------------
// Host launcher
// ---------------------------------------------------------------------------
extern "C" void kernel_launch(void* const* d_in, const int* in_sizes, int n_in,
                              void* d_out, int out_size, void* d_ws, size_t ws_size,
                              hipStream_t stream) {
  (void)in_sizes; (void)n_in; (void)out_size; (void)ws_size;

  const float* inputs   = (const float*)d_in[0];
  const float* targets  = (const float*)d_in[1];
  const float* We       = (const float*)d_in[2];
  const float* be       = (const float*)d_in[3];
  const float* Wa       = (const float*)d_in[4];
  const float* ba       = (const float*)d_in[5];
  const float* Ua       = (const float*)d_in[6];
  const float* bu       = (const float*)d_in[7];
  const float* Va       = (const float*)d_in[8];
  const float* bv       = (const float*)d_in[9];
  const float* enc_Wih  = (const float*)d_in[10];
  const float* enc_Whh  = (const float*)d_in[11];
  const float* enc_bih  = (const float*)d_in[12];
  const float* enc_bhh  = (const float*)d_in[13];
  const float* dec_Wih0 = (const float*)d_in[14];
  const float* dec_Wih  = (const float*)d_in[15];
  const float* dec_Whh  = (const float*)d_in[16];
  const float* dec_bih  = (const float*)d_in[17];
  const float* dec_bhh  = (const float*)d_in[18];

  // Workspace carving
  char* base = (char*)d_ws;
  size_t off = 0;
  auto carve = [&](size_t bytes) -> char* {
    char* p = base + off;
    off += (bytes + 255) & ~(size_t)255;
    return p;
  };
  unsigned* bar       = (unsigned*)carve(256);
  bf16_t* encWih_bf   = (bf16_t*)carve((size_t)NLay * Hdim * Hdim * 2);
  bf16_t* encWhh_bf   = (bf16_t*)carve((size_t)NLay * Hdim * Hdim * 2);
  bf16_t* Ua_bf       = (bf16_t*)carve((size_t)Hdim * Hdim * 2);
  bf16_t* Wa_bf       = (bf16_t*)carve((size_t)Hdim * Hdim * 2);
  bf16_t* Wih0_bf     = (bf16_t*)carve((size_t)Hdim * 2 * Hdim * 2);
  bf16_t* decWih_bf   = (bf16_t*)carve((size_t)(NLay - 1) * Hdim * Hdim * 2);
  bf16_t* decWhh_bf   = (bf16_t*)carve((size_t)NLay * Hdim * Hdim * 2);
  bf16_t* x_bf        = (bf16_t*)carve((size_t)Bsz * Llen * Hdim * 2);
  float*  xw          = (float*) carve((size_t)Bsz * Llen * Hdim * 4);
  float*  ukeys       = (float*) carve((size_t)Bsz * Llen * Hdim * 4);
  bf16_t* hping       = (bf16_t*)carve((size_t)Bsz * Hdim * 2);
  bf16_t* hpong       = (bf16_t*)carve((size_t)Bsz * Hdim * 2);
  float*  hidden_f32  = (float*) carve((size_t)NLay * Bsz * Hdim * 4);
  bf16_t* hidden_bf   = (bf16_t*)carve((size_t)NLay * Bsz * Hdim * 2);
  float*  w_f32       = (float*) carve((size_t)Bsz * Hdim * 4);
  float*  attn_ws     = (float*) carve((size_t)Bsz * Llen * 4);
  bf16_t* inp0_bf     = (bf16_t*)carve((size_t)Bsz * 2 * Hdim * 2);
  bf16_t* inpA_bf     = (bf16_t*)carve((size_t)Bsz * Hdim * 2);
  bf16_t* inpB_bf     = (bf16_t*)carve((size_t)Bsz * Hdim * 2);
  float*  top_f32     = (float*) carve((size_t)Bsz * Hdim * 4);

  float* out    = (float*)d_out;
  float* out_lp = out;
  float* out_hT = out + (size_t)Bsz * Tlen * Hdim;
  float* out_at = out_hT + (size_t)NLay * Bsz * Hdim;

  // 1) weight conversion to bf16
  auto conv = [&](const float* s, bf16_t* d, int n) {
    k_conv_bf16<<<(n + 255) / 256, 256, 0, stream>>>(s, d, n);
  };
  conv(enc_Wih,  encWih_bf, NLay * Hdim * Hdim);
  conv(enc_Whh,  encWhh_bf, NLay * Hdim * Hdim);
  conv(Ua,       Ua_bf,     Hdim * Hdim);
  conv(Wa,       Wa_bf,     Hdim * Hdim);
  conv(dec_Wih0, Wih0_bf,   Hdim * 2 * Hdim);
  conv(dec_Wih,  decWih_bf, (NLay - 1) * Hdim * Hdim);
  conv(dec_Whh,  decWhh_bf, NLay * Hdim * Hdim);

  // 2) embedding
  {
    int total = Bsz * Llen * Hdim;
    k_embed<<<(total + 255) / 256, 256, 0, stream>>>(inputs, We, be, x_bf, total);
  }

  // 3) encoder layers: big GEMM for xw, then persistent scan
  const int M = Bsz * Llen;
  const int gemm_blocks = ((M / 16) * (Hdim / 16) * 32 + 255) / 256;
  for (int l = 0; l < NLay; ++l) {
    k_gemm_bias<<<gemm_blocks, 256, 0, stream>>>(
        x_bf, encWih_bf + (size_t)l * Hdim * Hdim, enc_bih + (size_t)l * Hdim,
        xw, M, Hdim, Hdim);
    (void)hipMemsetAsync(bar, 0, 4, stream);
    (void)hipMemsetAsync(hping, 0, (size_t)Bsz * Hdim * 2, stream);
    k_enc_scan<<<ENC_BLOCKS, 256, 0, stream>>>(
        xw, x_bf, encWhh_bf + (size_t)l * Hdim * Hdim, enc_bhh + (size_t)l * Hdim,
        hping, hpong, hidden_f32 + (size_t)l * Bsz * Hdim, bar);
  }

  // 4) u_keys = enc_out @ Ua^T + bu  (f32, stays L2-resident for decode)
  k_gemm_bias<<<gemm_blocks, 256, 0, stream>>>(x_bf, Ua_bf, bu, ukeys, M, Hdim, Hdim);

  // 5) decoder hidden init (bf16 copy of encoder finals)
  conv(hidden_f32, hidden_bf, NLay * Bsz * Hdim);

  // 6) persistent decoder
  (void)hipMemsetAsync(bar, 0, 4, stream);
  DecParams p;
  p.targets = targets; p.We = We; p.be = be;
  p.Wa_bf = Wa_bf; p.ba = ba; p.Va = Va; p.bv = bv;
  p.ukeys = ukeys; p.enc_bf = x_bf;
  p.Wih0_bf = Wih0_bf; p.Wih_bf = decWih_bf; p.Whh_bf = decWhh_bf;
  p.bih = dec_bih; p.bhh = dec_bhh;
  p.hidden_f32 = hidden_f32; p.hidden_bf = hidden_bf;
  p.w_f32 = w_f32; p.attn_ws = attn_ws;
  p.inp0_bf = inp0_bf; p.inpA_bf = inpA_bf; p.inpB_bf = inpB_bf;
  p.top_f32 = top_f32;
  p.out_lp = out_lp; p.out_hT = out_hT; p.out_attn = out_at;
  p.bar = bar;
  k_decoder<<<DEC_BLOCKS, 256, 0, stream>>>(p);
}
